// LinearSSM_31516470018469
// MI455X (gfx1250) — matrix operand order repaired and checked by reading, not compiled, not run on hardware
//
#include <hip/hip_runtime.h>

// Problem constants (B,T,C) = (4, 4096, 2048)
#define BB     4
#define TT     4096
#define CC     2048
#define M_TOT  (BB * TT)   // 16384 rows of x (flattened b,t)
#define N_TOT  (2 * CC)    // 4096 output cols (gates || values)
#define K_TOT  CC          // 2048 reduction

// GEMM tiling
#define MT   128           // workgroup M tile
#define NT   64            // workgroup N tile
#define KC   32            // K chunk per LDS stage
#define LDA  (KC + 8)      // LDS row pitch in bf16 (80B -> conflict-free b128)

// Scan chunking
#define NCHUNK 32
#define CHLEN  (TT / NCHUNK)   // 128

typedef __attribute__((ext_vector_type(16))) __bf16 v16bf;
typedef __attribute__((ext_vector_type(8)))  __bf16 v8bf;
typedef __attribute__((ext_vector_type(8)))  float  v8f;
typedef __attribute__((ext_vector_type(4)))  float  vf4;

// Branch-free activations (v_exp_f32 + v_rcp_f32; no libm divergence)
__device__ __forceinline__ float sigmoid_fast(float x) {
  return __builtin_amdgcn_rcpf(1.0f + __expf(-x));
}
__device__ __forceinline__ float tanh_fast(float x) {
  const float t = __expf(-2.0f * fabsf(x));           // (0,1]
  const float r = (1.0f - t) * __builtin_amdgcn_rcpf(1.0f + t);
  return copysignf(r, x);
}

// ---------------------------------------------------------------------------
// GEMM (bf16x3 split precision) + bias + activation.
//   gbuf[m,c] = sigmoid(x@W^T + b)  for cols [0,C)
//   vbuf[m,c] = tanh   (x@W^T + b)  for cols [C,2C)
// ---------------------------------------------------------------------------
__global__ __launch_bounds__(256)
void gemm_act_kernel(const float* __restrict__ X, const float* __restrict__ W,
                     const float* __restrict__ bias,
                     float* __restrict__ gbuf, float* __restrict__ vbuf)
{
  __shared__ __align__(16) __bf16 As_hi[MT][LDA];
  __shared__ __align__(16) __bf16 As_lo[MT][LDA];
  __shared__ __align__(16) __bf16 Bs_hi[NT][LDA];
  __shared__ __align__(16) __bf16 Bs_lo[NT][LDA];

  const int tid  = threadIdx.x;
  const int lane = tid & 31;
  const int wid  = tid >> 5;
  const int wm   = wid & 3;       // wave M position: rows wm*32..+32
  const int wn   = wid >> 2;      // wave N position: cols wn*32..+32
  const int half = lane >> 4;     // lane half (WMMA layout)
  const int l15  = lane & 15;

  const int m0 = blockIdx.y * MT;
  const int n0 = blockIdx.x * NT;

  // global->LDS staging assignment
  const int arow = tid >> 1;      // 0..127, 16 floats each
  const int aseg = tid & 1;
  const int brow = tid >> 2;      // 0..63, 8 floats each
  const int bseg = tid & 3;

  const float* Xg = X + (size_t)(m0 + arow) * K_TOT + aseg * 16;
  const float* Wg = W + (size_t)(n0 + brow) * K_TOT + bseg * 8;

  v8f acc[2][2];
  #pragma unroll
  for (int i = 0; i < 2; ++i)
    #pragma unroll
    for (int j = 0; j < 2; ++j)
      #pragma unroll
      for (int e = 0; e < 8; ++e) acc[i][j][e] = 0.0f;

  // WMMA bf16 A-fragment: lanes 0-15 hold K{0-7,16-23}, lanes 16-31 K{8-15,24-31}
  const int khA = half * 8;
  // WMMA bf16 B-fragment: lanes 0-15 hold K{0-7,16-23}, lanes 16-31 K{8-15,24-31} (col = l15)
  const int krB = half * 8;

  // ---- prologue: load first K tile ----
  vf4 ar[4], br[2];
  {
    const vf4* xp = (const vf4*)Xg;
    ar[0] = xp[0]; ar[1] = xp[1]; ar[2] = xp[2]; ar[3] = xp[3];
    const vf4* wp = (const vf4*)Wg;
    br[0] = wp[0]; br[1] = wp[1];
  }

  for (int k0 = 0; k0 < K_TOT; k0 += KC) {
    __syncthreads();   // previous iteration's LDS reads complete

    // ---- split fp32 -> bf16 hi/lo, stage to LDS (ds_store_b128) ----
    #pragma unroll
    for (int q = 0; q < 2; ++q) {
      v8bf h, l;
      #pragma unroll
      for (int s = 0; s < 8; ++s) {
        float f   = ar[q * 2 + (s >> 2)][s & 3];
        __bf16 fh = (__bf16)f;
        h[s] = fh;
        l[s] = (__bf16)(f - (float)fh);
      }
      *(v8bf*)&As_hi[arow][aseg * 16 + q * 8] = h;
      *(v8bf*)&As_lo[arow][aseg * 16 + q * 8] = l;
    }
    {
      v8bf h, l;
      #pragma unroll
      for (int s = 0; s < 8; ++s) {
        float f   = br[s >> 2][s & 3];
        __bf16 fh = (__bf16)f;
        h[s] = fh;
        l[s] = (__bf16)(f - (float)fh);
      }
      *(v8bf*)&Bs_hi[brow][bseg * 8] = h;
      *(v8bf*)&Bs_lo[brow][bseg * 8] = l;
    }

    __syncthreads();

    // ---- software pipeline: unconditional next-tile loads (scalar-clamped K
    //      offset; last iteration harmlessly reloads the current tile from L2)
    //      issued here so they stream under the WMMA block below ----
    const int kn = (k0 + KC < K_TOT) ? (k0 + KC) : k0;   // s_cselect, uniform
    {
      const vf4* xp = (const vf4*)(Xg + kn);
      ar[0] = xp[0]; ar[1] = xp[1]; ar[2] = xp[2]; ar[3] = xp[3];
      const vf4* wp = (const vf4*)(Wg + kn);
      br[0] = wp[0]; br[1] = wp[1];
    }
    if (k0 + 2 * KC < K_TOT) {          // warm L2 at distance 2
      __builtin_prefetch(Xg + k0 + 2 * KC, 0, 1);
      __builtin_prefetch(Wg + k0 + 2 * KC, 0, 1);
    }

    // ---- load fragments from LDS ----
    v16bf ah[2], al[2], bh[2], bl[2];
    #pragma unroll
    for (int i = 0; i < 2; ++i) {
      const int r = wm * 32 + i * 16 + l15;
      v8bf x0 = *(const v8bf*)&As_hi[r][khA];
      v8bf x1 = *(const v8bf*)&As_hi[r][khA + 16];
      ah[i] = __builtin_shufflevector(x0, x1, 0,1,2,3,4,5,6,7,8,9,10,11,12,13,14,15);
      v8bf y0 = *(const v8bf*)&As_lo[r][khA];
      v8bf y1 = *(const v8bf*)&As_lo[r][khA + 16];
      al[i] = __builtin_shufflevector(y0, y1, 0,1,2,3,4,5,6,7,8,9,10,11,12,13,14,15);
    }
    #pragma unroll
    for (int j = 0; j < 2; ++j) {
      const int n = wn * 32 + j * 16 + l15;
      v8bf x0 = *(const v8bf*)&Bs_hi[n][krB];
      v8bf x1 = *(const v8bf*)&Bs_hi[n][krB + 16];
      bh[j] = __builtin_shufflevector(x0, x1, 0,1,2,3,4,5,6,7,8,9,10,11,12,13,14,15);
      v8bf y0 = *(const v8bf*)&Bs_lo[n][krB];
      v8bf y1 = *(const v8bf*)&Bs_lo[n][krB + 16];
      bl[j] = __builtin_shufflevector(y0, y1, 0,1,2,3,4,5,6,7,8,9,10,11,12,13,14,15);
    }

    // ---- bf16x3 WMMA, pass-major so accumulator reuse is 4 WMMAs apart ----
    #pragma unroll
    for (int i = 0; i < 2; ++i)
      #pragma unroll
      for (int j = 0; j < 2; ++j)
        acc[i][j] = __builtin_amdgcn_wmma_f32_16x16x32_bf16(
            false, ah[i], false, bh[j], (short)0, acc[i][j], false, false);
    #pragma unroll
    for (int i = 0; i < 2; ++i)
      #pragma unroll
      for (int j = 0; j < 2; ++j)
        acc[i][j] = __builtin_amdgcn_wmma_f32_16x16x32_bf16(
            false, ah[i], false, bl[j], (short)0, acc[i][j], false, false);
    #pragma unroll
    for (int i = 0; i < 2; ++i)
      #pragma unroll
      for (int j = 0; j < 2; ++j)
        acc[i][j] = __builtin_amdgcn_wmma_f32_16x16x32_bf16(
            false, al[i], false, bh[j], (short)0, acc[i][j], false, false);
  }

  // ---- epilogue: bias + branch-free activation ----
  const bool is_gate = (n0 < CC);   // uniform per block (NT=64 divides CC)
  #pragma unroll
  for (int i = 0; i < 2; ++i) {
    const int mg = m0 + wm * 32 + i * 16 + half * 8;  // lane holds rows mg..mg+7
    #pragma unroll
    for (int j = 0; j < 2; ++j) {
      const int ng = n0 + wn * 32 + j * 16 + l15;
      const float bs = bias[ng];
      #pragma unroll
      for (int r = 0; r < 8; ++r) {
        const float val = acc[i][j][r] + bs;
        if (is_gate) {
          gbuf[(size_t)(mg + r) * CC + ng] = sigmoid_fast(val);
        } else {
          vbuf[(size_t)(mg + r) * CC + (ng - CC)] = tanh_fast(val);
        }
      }
    }
  }
}

// ---------------------------------------------------------------------------
// Scan phase 1: per (b, chunk, c) compute chunk composition
//   G = prod g_t,  H = scan(chunk | h_in = 0) final value
// ---------------------------------------------------------------------------
__global__ __launch_bounds__(256)
void scan_phase1(const float* __restrict__ g, const float* __restrict__ v,
                 float* __restrict__ Gsum, float* __restrict__ Hsum)
{
  const int tid   = blockIdx.x * blockDim.x + threadIdx.x;  // 0..262143
  const int c     = tid & (CC - 1);
  const int bc    = tid >> 11;         // b*NCHUNK + chunk, 0..127
  const int b     = bc >> 5;
  const int chunk = bc & (NCHUNK - 1);

  const size_t base = ((size_t)b * TT + (size_t)chunk * CHLEN) * CC + c;

  float G = 1.0f, H = 0.0f;
  #pragma unroll 8
  for (int t = 0; t < CHLEN; ++t) {
    const float gt = g[base + (size_t)t * CC];
    const float vt = v[base + (size_t)t * CC];
    H = fmaf(gt, H, vt);
    G = G * gt;
  }
  Gsum[(size_t)bc * CC + c] = G;
  Hsum[(size_t)bc * CC + c] = H;
}

// ---------------------------------------------------------------------------
// Scan phase 2: fold preceding chunk summaries into a carry, replay chunk,
// writing h over v in d_out (in-place safe: read-before-write per element).
// ---------------------------------------------------------------------------
__global__ __launch_bounds__(256)
void scan_phase2(const float* __restrict__ g, const float* __restrict__ v,
                 const float* __restrict__ Gsum, const float* __restrict__ Hsum,
                 float* __restrict__ out)
{
  const int tid   = blockIdx.x * blockDim.x + threadIdx.x;
  const int c     = tid & (CC - 1);
  const int bc    = tid >> 11;
  const int b     = bc >> 5;
  const int chunk = bc & (NCHUNK - 1);

  float carry = 0.0f;
  for (int j = 0; j < chunk; ++j) {           // uniform per wave: no divergence
    const size_t s = ((size_t)(b * NCHUNK + j)) * CC + c;
    carry = fmaf(Gsum[s], carry, Hsum[s]);
  }

  const size_t base = ((size_t)b * TT + (size_t)chunk * CHLEN) * CC + c;
  float h = carry;
  #pragma unroll 8
  for (int t = 0; t < CHLEN; ++t) {
    const size_t off = base + (size_t)t * CC;
    const float gt = g[off];
    const float vt = v[off];
    h = fmaf(gt, h, vt);
    out[off] = h;
  }
}

// ---------------------------------------------------------------------------
extern "C" void kernel_launch(void* const* d_in, const int* in_sizes, int n_in,
                              void* d_out, int out_size, void* d_ws, size_t ws_size,
                              hipStream_t stream) {
  (void)in_sizes; (void)n_in; (void)out_size; (void)ws_size;

  const float* X = (const float*)d_in[0];   // [B,T,C]
  const float* W = (const float*)d_in[1];   // [2C,C]
  const float* b = (const float*)d_in[2];   // [2C]

  float* vbuf = (float*)d_out;                       // tanh values, then h in place
  float* gbuf = (float*)d_ws;                        // sigmoid gates (134 MB)
  float* Gsum = gbuf + (size_t)M_TOT * CC;           // [B*NCHUNK, C]
  float* Hsum = Gsum + (size_t)BB * NCHUNK * CC;     // [B*NCHUNK, C]

  dim3 grid(N_TOT / NT, M_TOT / MT);   // (64, 128)
  gemm_act_kernel<<<grid, 256, 0, stream>>>(X, W, b, gbuf, vbuf);

  const int nscan = BB * NCHUNK * CC;  // 262144 threads
  scan_phase1<<<nscan / 256, 256, 0, stream>>>(gbuf, vbuf, Gsum, Hsum);
  scan_phase2<<<nscan / 256, 256, 0, stream>>>(gbuf, vbuf, Gsum, Hsum, (float*)d_out);
}
